// DeepseekV4HybridLayer_36472862277795
// MI455X (gfx1250) — matrix operand order repaired and checked by reading, not compiled, not run on hardware
//
#include <hip/hip_runtime.h>

typedef __bf16 bf16;
typedef __attribute__((ext_vector_type(16))) __bf16 v16bf;
typedef __attribute__((ext_vector_type(8)))  __bf16 v8bf;
typedef __attribute__((ext_vector_type(8)))  float  v8f;

#define BB    2
#define SS    2048
#define DDIM  2048
#define HH    16
#define KVHH  4
#define HDD   128
#define FFND  8192
#define MTOK  4096          /* B*S */
#define NQKV  3072          /* H*HD + 2*KVH*HD */
#define WIN   1024
#define SCALE 0.08838834764831843f  /* 1/sqrt(128) */

__device__ __forceinline__ v8f wmma_bf16(v16bf a, v16bf b, v8f c) {
  return __builtin_amdgcn_wmma_f32_16x16x32_bf16(false, a, false, b, (short)0, c, false, false);
}

__device__ __forceinline__ v16bf ld_frag(const bf16* p0, const bf16* p1) {
  union { uint4 u[2]; v16bf v; } t;
  t.u[0] = *(const uint4*)p0;
  t.u[1] = *(const uint4*)p1;
  return t.v;
}

// ---- CDNA5 async global->LDS copy (ASYNCcnt-tracked), GVS addressing ----
__device__ __forceinline__ unsigned lds_addr_of(const void* p) {
  return (unsigned)(unsigned long long)p;            // low 32 bits = LDS offset
}
__device__ __forceinline__ void async_ld_b128(unsigned lds_addr, unsigned goff,
                                              const void* base) {
  unsigned long long sbase = (unsigned long long)base;
  asm volatile("global_load_async_to_lds_b128 %0, %1, %2"
               :: "v"(lds_addr), "v"(goff), "s"(sbase)
               : "memory");
}
__device__ __forceinline__ void wait_async0() {
  asm volatile("s_wait_asynccnt 0x0" ::: "memory");
}

// ---------------------------------------------------------------------------
// fp32 -> bf16 weight conversion / QKV weight packing
// ---------------------------------------------------------------------------
__global__ void cvt_bf16_kernel(const float* __restrict__ in, bf16* __restrict__ out, int n) {
  int i = blockIdx.x * 256 + threadIdx.x;
  if (i < n) out[i] = (bf16)in[i];
}

__global__ void pack_qkv_kernel(const float* __restrict__ wq, const float* __restrict__ wk,
                                const float* __restrict__ wv, bf16* __restrict__ out) {
  int i = blockIdx.x * 256 + threadIdx.x;            // over D*NQKV = 2048*3072
  int r = i / NQKV, c = i - r * NQKV;
  float v;
  if (c < 2048)        v = wq[(size_t)r * 2048 + c];
  else if (c < 2560)   v = wk[(size_t)r * 512 + (c - 2048)];
  else                 v = wv[(size_t)r * 512 + (c - 2560)];
  out[i] = (bf16)v;
}

// ---------------------------------------------------------------------------
// RMSNorm: fp32 in -> bf16 out  (one block per row, D = 2048)
// ---------------------------------------------------------------------------
__global__ __launch_bounds__(256) void rmsnorm_kernel(const float* __restrict__ x,
                                                      const float* __restrict__ w,
                                                      bf16* __restrict__ out) {
  const int row = blockIdx.x;
  const float* xr = x + (size_t)row * DDIM;
  float s = 0.f;
  for (int i = threadIdx.x; i < DDIM; i += 256) { float v = xr[i]; s += v * v; }
  __shared__ float red[256];
  red[threadIdx.x] = s;
  __syncthreads();
  for (int o = 128; o > 0; o >>= 1) {
    if (threadIdx.x < o) red[threadIdx.x] += red[threadIdx.x + o];
    __syncthreads();
  }
  float r = rsqrtf(red[0] * (1.f / DDIM) + 1e-6f);
  for (int i = threadIdx.x; i < DDIM; i += 256)
    out[(size_t)row * DDIM + i] = (bf16)(xr[i] * r * w[i]);
}

// ---------------------------------------------------------------------------
// RoPE over packed qkv[M, 3072] (bf16, in place); RD=64 -> 32 pairs/head
// ---------------------------------------------------------------------------
__global__ void rope_kernel(bf16* __restrict__ qkv) {
  const int PPR = (HH + KVHH) * 32;                  // 640 pairs per token row
  int idx = blockIdx.x * 256 + threadIdx.x;
  if (idx >= MTOK * PPR) return;
  int row = idx / PPR;
  int rem = idx - row * PPR;
  int hp  = rem >> 5;
  int i   = rem & 31;
  int s   = row & (SS - 1);
  int colb = (hp < HH) ? hp * HDD : 2048 + (hp - HH) * HDD;
  size_t base = (size_t)row * NQKV + colb + 2 * i;
  float inv_freq = __expf(-(float)(2 * i) * (1.f / 64.f) * __logf(10000.f));
  float ang = (float)s * inv_freq;
  float sn, cs;
  __sincosf(ang, &sn, &cs);
  float r1 = (float)qkv[base], r2 = (float)qkv[base + 1];
  qkv[base]     = (bf16)(r1 * cs - r2 * sn);
  qkv[base + 1] = (bf16)(r1 * sn + r2 * cs);
}

// ---------------------------------------------------------------------------
// BF16 WMMA GEMM: C[M,N] = A[M,K] x B[K,N]  (+ epilogue)
// block tile 128x128, K-step 32, 8 waves: 4(M) x 2(N), wave tile 32x64
// A tile staged via async global->LDS; B staged transposed via VGPRs.
// ---------------------------------------------------------------------------
#define EPI_BF16 0
#define EPI_RES  1
#define EPI_SILU 2
#define EPI_MUL  3

template <int EPI>
__global__ __launch_bounds__(256) void gemm_bf16_kernel(const bf16* __restrict__ A,
                                                        const bf16* __restrict__ Bm,
                                                        void* __restrict__ Cv,
                                                        const float* __restrict__ R,
                                                        int N, int K) {
  __shared__ __align__(16) bf16 sA[2][128 * 32];
  __shared__ __align__(16) bf16 sB[2][128 * 40];    // transposed [n][k], pitch 40

  const int t    = threadIdx.x;
  const int lane = t & 31;
  const int wave = t >> 5;
  const int wm   = wave >> 1;                       // 0..3
  const int wn   = wave & 1;                        // 0..1
  const int hl   = lane >> 4;
  const int ml   = lane & 15;

  const int bn0 = blockIdx.x * 128;
  const int bm0 = blockIdx.y * 128;

  const int ra = t >> 2, ca = (t & 3) * 8;          // A: two rows ra, ra+64
  const int rb = t >> 4, cb = (t & 15) * 8;         // B: two rows rb, rb+16

  const unsigned ldsA0 = lds_addr_of(&sA[0][ra * 32 + ca]);
  const unsigned ldsA1 = lds_addr_of(&sA[0][(ra + 64) * 32 + ca]);
  const unsigned ldsAstride = (unsigned)(128 * 32 * 2);   // bytes between buffers

  v8f zero = {};
  v8f acc[2][4];
#pragma unroll
  for (int i = 0; i < 2; ++i)
#pragma unroll
    for (int j = 0; j < 4; ++j) acc[i][j] = zero;

  const int NT = K >> 5;

  {
    async_ld_b128(ldsA0, (unsigned)(((size_t)(bm0 + ra) * K + ca) * 2), A);
    async_ld_b128(ldsA1, (unsigned)(((size_t)(bm0 + ra + 64) * K + ca) * 2), A);
    v8bf b0 = *(const v8bf*)(Bm + (size_t)rb * N + bn0 + cb);
    v8bf b1 = *(const v8bf*)(Bm + (size_t)(rb + 16) * N + bn0 + cb);
#pragma unroll
    for (int i = 0; i < 8; ++i) {
      sB[0][(cb + i) * 40 + rb]      = b0[i];
      sB[0][(cb + i) * 40 + rb + 16] = b1[i];
    }
    wait_async0();
  }
  __syncthreads();

  for (int kt = 0; kt < NT; ++kt) {
    const int cur = kt & 1;
    v8bf b0, b1;
    if (kt + 1 < NT) {
      const int k0 = (kt + 1) << 5;
      const unsigned boff = (cur ^ 1) ? ldsAstride : 0u;
      async_ld_b128(ldsA0 + boff, (unsigned)(((size_t)(bm0 + ra) * K + k0 + ca) * 2), A);
      async_ld_b128(ldsA1 + boff, (unsigned)(((size_t)(bm0 + ra + 64) * K + k0 + ca) * 2), A);
      b0 = *(const v8bf*)(Bm + (size_t)(k0 + rb) * N + bn0 + cb);
      b1 = *(const v8bf*)(Bm + (size_t)(k0 + rb + 16) * N + bn0 + cb);
      if (kt + 2 < NT)
        __builtin_prefetch(A + (size_t)(bm0 + ra) * K + ((kt + 2) << 5) + ca, 0, 0);
    }

    v16bf af[2], bfr[4];
#pragma unroll
    for (int fm = 0; fm < 2; ++fm) {
      const bf16* p = &sA[cur][(wm * 32 + fm * 16 + ml) * 32];
      af[fm] = ld_frag(p + hl * 8, p + 16 + hl * 8);
    }
#pragma unroll
    for (int fn = 0; fn < 4; ++fn) {
      const bf16* p = &sB[cur][(wn * 64 + fn * 16 + ml) * 40 + hl * 8];
      bfr[fn] = ld_frag(p, p + 16);
    }
#pragma unroll
    for (int fm = 0; fm < 2; ++fm)
#pragma unroll
      for (int fn = 0; fn < 4; ++fn)
        acc[fm][fn] = wmma_bf16(af[fm], bfr[fn], acc[fm][fn]);

    if (kt + 1 < NT) {
      const int nxt = cur ^ 1;
#pragma unroll
      for (int i = 0; i < 8; ++i) {
        sB[nxt][(cb + i) * 40 + rb]      = b0[i];
        sB[nxt][(cb + i) * 40 + rb + 16] = b1[i];
      }
      wait_async0();
    }
    __syncthreads();
  }

  (void)R;
#pragma unroll
  for (int fm = 0; fm < 2; ++fm)
#pragma unroll
    for (int fn = 0; fn < 4; ++fn)
#pragma unroll
      for (int g = 0; g < 8; ++g) {
        const int row = bm0 + wm * 32 + fm * 16 + hl * 8 + g;
        const int col = bn0 + wn * 64 + fn * 16 + ml;
        const size_t o = (size_t)row * N + col;
        const float v = acc[fm][fn][g];
        if constexpr (EPI == EPI_BF16) {
          ((bf16*)Cv)[o] = (bf16)v;
        } else if constexpr (EPI == EPI_RES) {
          ((float*)Cv)[o] = v + R[o];
        } else if constexpr (EPI == EPI_SILU) {
          ((bf16*)Cv)[o] = (bf16)(v / (1.f + __expf(-v)));
        } else {
          bf16* C = (bf16*)Cv;
          C[o] = (bf16)(v * (float)C[o]);
        }
      }
}

// ---------------------------------------------------------------------------
// Sliding-window causal flash attention, GQA-cooperative.
// One block = (b, kv-head, 32-query chunk); 8 waves = 4 Q-heads x 2 q-blocks.
// K staged via async global->LDS ([key][d]); V transposed via VGPRs ([d][key]).
// Interior key-pairs (fully inside the window for every lane) skip all
// mask compares / cndmasks / exp guards.
// ---------------------------------------------------------------------------
__global__ __launch_bounds__(256) void attn_kernel(const bf16* __restrict__ qkv,
                                                   bf16* __restrict__ out) {
  __shared__ __align__(16) bf16 sK[2][32 * 128];    // [key][d]      8 KB/buf
  __shared__ __align__(16) bf16 sV[2][128 * 40];    // [d][key] p40  10 KB/buf

  const int t    = threadIdx.x;
  const int lane = t & 31;
  const int wave = t >> 5;

  const int blk    = blockIdx.x;                    // 512 = B * KVH * (S/32)
  const int qchunk = blk & 63;
  const int kvh    = (blk >> 6) & 3;
  const int b      = blk >> 8;
  const int qb0    = qchunk * 2;
  const int h      = kvh * 4 + (wave & 3);
  const int qb     = qb0 + (wave >> 2);

  const int hl = lane >> 4;
  const int n  = lane & 15;
  const int q  = qb * 16 + n;                       // this lane's query column

  const int rr = t >> 4;                            // 0..15 (rows rr, rr+16)
  const int cc = (t & 15) * 8;                      // d-offset 0..120

  const size_t kcol0 = 2048 + kvh * HDD;
  const size_t vcol0 = 2560 + kvh * HDD;

  const unsigned ldsK0 = lds_addr_of(&sK[0][rr * 128 + cc]);
  const unsigned ldsK1 = lds_addr_of(&sK[0][(rr + 16) * 128 + cc]);
  const unsigned ldsKstride = (unsigned)(32 * 128 * 2);

  // Q^T B-fragments (4 chunks of 32 along head dim)
  v16bf bq[4];
  {
    const bf16* qrow = qkv + (size_t)(b * SS + q) * NQKV + h * HDD;
#pragma unroll
    for (int c = 0; c < 4; ++c) {
      const bf16* p = qrow + c * 32 + hl * 8;
      bq[c] = ld_frag(p, p + 16);
    }
  }

  v8f zero = {};
  v8f acc[8];
#pragma unroll
  for (int fd = 0; fd < 8; ++fd) acc[fd] = zero;
  float m_run = -1e30f, l_run = 0.f;

  const int lo    = (qb0 >= 64) ? (qb0 - 64) : 0;   // union window start block
  const int hi    = qb0 + 1;                        // last key block
  const int npair = (hi - lo + 2) >> 1;

  v8bf vr0, vr1;
  auto load_pair = [&](int kb, int buf) {
    int key0 = kb * 16 + rr;       if (key0 > SS - 1) key0 = SS - 1;
    int key1 = kb * 16 + rr + 16;  if (key1 > SS - 1) key1 = SS - 1;
    const unsigned boff = buf ? ldsKstride : 0u;
    async_ld_b128(ldsK0 + boff, (unsigned)(((size_t)(b * SS + key0) * NQKV + kcol0 + cc) * 2), qkv);
    async_ld_b128(ldsK1 + boff, (unsigned)(((size_t)(b * SS + key1) * NQKV + kcol0 + cc) * 2), qkv);
    vr0 = *(const v8bf*)(qkv + (size_t)(b * SS + key0) * NQKV + vcol0 + cc);
    vr1 = *(const v8bf*)(qkv + (size_t)(b * SS + key1) * NQKV + vcol0 + cc);
  };
  auto store_v = [&](int buf) {
#pragma unroll
    for (int i = 0; i < 8; ++i) {
      sV[buf][(cc + i) * 40 + rr]      = vr0[i];
      sV[buf][(cc + i) * 40 + rr + 16] = vr1[i];
    }
  };

  load_pair(lo, 0);
  store_v(0);
  wait_async0();
  __syncthreads();

  for (int it = 0; it < npair; ++it) {
    const int kb  = lo + it * 2;
    const int cur = it & 1;
    if (it + 1 < npair) load_pair(kb + 2, cur ^ 1);

    // --- scores for the two 16-key blocks of this pair ---
    v8f st0 = zero, st1 = zero;
#pragma unroll
    for (int c = 0; c < 4; ++c) {
      const bf16* p0 = &sK[cur][n * 128 + c * 32];
      st0 = wmma_bf16(ld_frag(p0 + hl * 8, p0 + 16 + hl * 8), bq[c], st0);
    }
#pragma unroll
    for (int c = 0; c < 4; ++c) {
      const bf16* p1 = &sK[cur][(16 + n) * 128 + c * 32];
      st1 = wmma_bf16(ld_frag(p1 + hl * 8, p1 + 16 + hl * 8), bq[c], st1);
    }

    // block-uniform: pair fully valid for every lane of every wave?
    const bool full = (kb >= qb0 - 62) && (kb <= qb0 - 2);

    float s0[8], s1[8];
    if (full) {
#pragma unroll
      for (int g = 0; g < 8; ++g) { s0[g] = st0[g] * SCALE; s1[g] = st1[g] * SCALE; }
    } else {
#pragma unroll
      for (int g = 0; g < 8; ++g) {
        int k0 = kb * 16 + hl * 8 + g;
        bool v0 = (k0 <= q) && ((q - k0) < WIN);
        s0[g] = v0 ? st0[g] * SCALE : -1e30f;
        int k1 = (kb + 1) * 16 + hl * 8 + g;
        bool v1 = (k1 <= q) && ((q - k1) < WIN);
        s1[g] = v1 ? st1[g] * SCALE : -1e30f;
      }
    }

    // --- online softmax (per-lane stats; halves paired via shfl_xor 16) ---
    float mloc = -1e30f;
#pragma unroll
    for (int g = 0; g < 8; ++g) mloc = fmaxf(mloc, fmaxf(s0[g], s1[g]));
    mloc = fmaxf(mloc, __shfl_xor(mloc, 16, 32));
    const float m_new = fmaxf(m_run, mloc);
    const float alpha = __expf(m_run - m_new);

    float e0[8], e1[8], lad = 0.f;
    if (full) {
#pragma unroll
      for (int g = 0; g < 8; ++g) {
        e0[g] = __expf(s0[g] - m_new);
        e1[g] = __expf(s1[g] - m_new);
        lad += e0[g] + e1[g];
      }
    } else {
#pragma unroll
      for (int g = 0; g < 8; ++g) {
        e0[g] = (s0[g] < -1e29f) ? 0.f : __expf(s0[g] - m_new);  // guard all-masked
        e1[g] = (s1[g] < -1e29f) ? 0.f : __expf(s1[g] - m_new);
        lad += e0[g] + e1[g];
      }
    }
    lad += __shfl_xor(lad, 16, 32);
    l_run = l_run * alpha + lad;
    m_run = m_new;
#pragma unroll
    for (int fd = 0; fd < 8; ++fd)
#pragma unroll
      for (int g = 0; g < 8; ++g) acc[fd][g] *= alpha;

    // --- P^T (C-layout f32) -> B-fragment (bf16) via shfl_xor(16) ---
    v16bf pb;
#pragma unroll
    for (int g = 0; g < 8; ++g) {
      float o0 = __shfl_xor(e0[g], 16, 32);
      float o1 = __shfl_xor(e1[g], 16, 32);
      pb[g]     = (bf16)(hl ? o1 : e0[g]);
      pb[g + 8] = (bf16)(hl ? e1[g] : o0);
    }

    // --- O^T += V^T x P^T  (A = V^T from transposed LDS; 2x b128/frag) ---
#pragma unroll
    for (int fd = 0; fd < 8; ++fd) {
      const bf16* vp = &sV[cur][(fd * 16 + n) * 40];
      acc[fd] = wmma_bf16(ld_frag(vp + hl * 16, vp + hl * 16 + 8), pb, acc[fd]);
    }

    if (it + 1 < npair) {
      store_v(cur ^ 1);
      wait_async0();
    }
    __syncthreads();
  }

  // --- normalize + store O (row = token, col = h*128 + d), packed b128 ---
  const float inv = 1.f / l_run;
  bf16* orow = out + (size_t)(b * SS + q) * DDIM + h * HDD;
#pragma unroll
  for (int fd = 0; fd < 8; ++fd) {
    v8bf ov;
#pragma unroll
    for (int g = 0; g < 8; ++g) ov[g] = (bf16)(acc[fd][g] * inv);
    *(v8bf*)(orow + fd * 16 + hl * 8) = ov;
  }
}

// ---------------------------------------------------------------------------
// host-side launcher
// ---------------------------------------------------------------------------
extern "C" void kernel_launch(void* const* d_in, const int* in_sizes, int n_in,
                              void* d_out, int out_size, void* d_ws, size_t ws_size,
                              hipStream_t stream) {
  (void)in_sizes; (void)n_in; (void)out_size; (void)ws_size;
  const float* x       = (const float*)d_in[0];
  const float* attn_w  = (const float*)d_in[2];
  const float* wq      = (const float*)d_in[3];
  const float* wk      = (const float*)d_in[4];
  const float* wv      = (const float*)d_in[5];
  const float* wo      = (const float*)d_in[6];
  const float* ffn_w   = (const float*)d_in[7];
  const float* w_gate  = (const float*)d_in[8];
  const float* w_up    = (const float*)d_in[9];
  const float* w_down  = (const float*)d_in[10];

  char* p = (char*)d_ws;
  auto alloc = [&](size_t bytes) { void* r = p; p += (bytes + 255) & ~(size_t)255; return r; };
  bf16*  wqkv_b  = (bf16*)alloc((size_t)DDIM * NQKV * 2);
  bf16*  wo_b    = (bf16*)alloc((size_t)DDIM * DDIM * 2);
  bf16*  wgate_b = (bf16*)alloc((size_t)DDIM * FFND * 2);
  bf16*  wup_b   = (bf16*)alloc((size_t)DDIM * FFND * 2);
  bf16*  wdown_b = (bf16*)alloc((size_t)FFND * DDIM * 2);
  bf16*  h_b     = (bf16*)alloc((size_t)MTOK * DDIM * 2);
  bf16*  qkv_b   = (bf16*)alloc((size_t)MTOK * NQKV * 2);
  bf16*  attn_b  = (bf16*)alloc((size_t)MTOK * DDIM * 2);
  bf16*  g_b     = (bf16*)alloc((size_t)MTOK * DDIM * 2);
  bf16*  hffn_b  = (bf16*)alloc((size_t)MTOK * FFND * 2);
  float* xmid    = (float*)alloc((size_t)MTOK * DDIM * 4);

  // 1) weight conversion / packing
  pack_qkv_kernel<<<(DDIM * NQKV) / 256, 256, 0, stream>>>(wq, wk, wv, wqkv_b);
  cvt_bf16_kernel<<<(DDIM * DDIM + 255) / 256, 256, 0, stream>>>(wo, wo_b, DDIM * DDIM);
  cvt_bf16_kernel<<<(DDIM * FFND + 255) / 256, 256, 0, stream>>>(w_gate, wgate_b, DDIM * FFND);
  cvt_bf16_kernel<<<(DDIM * FFND + 255) / 256, 256, 0, stream>>>(w_up, wup_b, DDIM * FFND);
  cvt_bf16_kernel<<<(FFND * DDIM + 255) / 256, 256, 0, stream>>>(w_down, wdown_b, FFND * DDIM);

  // 2) attn pre-norm, fused QKV projection, RoPE
  rmsnorm_kernel<<<MTOK, 256, 0, stream>>>(x, attn_w, h_b);
  gemm_bf16_kernel<EPI_BF16><<<dim3(NQKV / 128, MTOK / 128), 256, 0, stream>>>(
      h_b, wqkv_b, qkv_b, nullptr, NQKV, DDIM);
  rope_kernel<<<(MTOK * (HH + KVHH) * 32) / 256, 256, 0, stream>>>(qkv_b);

  // 3) sliding-window attention + O-proj with residual (x + o -> xmid)
  attn_kernel<<<BB * KVHH * (SS / 32), 256, 0, stream>>>(qkv_b, attn_b);
  gemm_bf16_kernel<EPI_RES><<<dim3(DDIM / 128, MTOK / 128), 256, 0, stream>>>(
      attn_b, wo_b, xmid, x, DDIM, DDIM);

  // 4) FFN pre-norm, SwiGLU (silu(gate) * up), down-proj with residual -> out
  rmsnorm_kernel<<<MTOK, 256, 0, stream>>>(xmid, ffn_w, g_b);
  gemm_bf16_kernel<EPI_SILU><<<dim3(FFND / 128, MTOK / 128), 256, 0, stream>>>(
      g_b, wgate_b, hffn_b, nullptr, FFND, DDIM);
  gemm_bf16_kernel<EPI_MUL><<<dim3(FFND / 128, MTOK / 128), 256, 0, stream>>>(
      g_b, wup_b, hffn_b, nullptr, FFND, DDIM);
  gemm_bf16_kernel<EPI_RES><<<dim3(DDIM / 128, MTOK / 128), 256, 0, stream>>>(
      hffn_b, wdown_b, (float*)d_out, xmid, DDIM, FFND);
}
